// SS2D_40776419508459
// MI455X (gfx1250) — matrix operand order repaired and checked by reading, not compiled, not run on hardware
//
#include <hip/hip_runtime.h>
#include <hip/hip_bf16.h>
#include <math.h>

// ---------------------------------------------------------------------------
// SS2D block for MI455X (gfx1250, wave32).
// GEMMs use v_wmma_f32_16x16x32_f16 (f16 A/B, f32 accumulate).
// c1/c3 GEMMs stage B tiles in LDS via GLOBAL_LOAD_ASYNC_TO_LDS_B128
// (ASYNCcnt-tracked, double-buffered). Scan uses global_prefetch.
// The 4 scan directions are index permutations of L; gathers are folded into
// the WMMA B-fragment loads and the scan, so xs is never materialized.
// ---------------------------------------------------------------------------

typedef __attribute__((ext_vector_type(16))) _Float16 v16h;
typedef __attribute__((ext_vector_type(8)))  float    v8f;
typedef __attribute__((ext_vector_type(4)))  float    f32x4;
typedef __attribute__((ext_vector_type(4)))  int      i32x4;

#define DEV __device__ __forceinline__
#define GLOBAL_AS __attribute__((address_space(1)))
#define LDS_AS    __attribute__((address_space(3)))

#if defined(__has_builtin)
#  if __has_builtin(__builtin_amdgcn_global_load_async_to_lds_b128)
#    define ASYNC_LDS_OK 1
#  endif
#endif
#ifndef ASYNC_LDS_OK
#  define ASYNC_LDS_OK 0
#endif

constexpr int B_ = 2, H_ = 48, W_ = 48, L_ = H_ * W_;      // L = 2304
constexpr int DM_ = 96, E_ = 192, E2_ = 384, R_ = 6, N_ = 16;
constexpr int EH_ = 96, EQ_ = 48;                           // E/2, E/4
constexpr int DROWS_ = R_ + 2 * N_;                         // 38 (padded to 48)

// workspace layout (fp32 elements)
constexpr size_t SZ_XZ  = (size_t)B_ * E2_ * L_;
constexpr size_t SZ_XI  = (size_t)B_ * E_  * L_;
constexpr size_t SZ_DBC = (size_t)B_ * 4 * 48 * L_;
constexpr size_t SZ_DLT = (size_t)B_ * 4 * E_ * L_;
constexpr size_t OFF_XZ  = 0;
constexpr size_t OFF_XI  = OFF_XZ  + SZ_XZ;
constexpr size_t OFF_DBC = OFF_XI  + SZ_XI;
constexpr size_t OFF_DLT = OFF_DBC + SZ_DBC;
constexpr size_t OFF_YB  = OFF_DLT + SZ_DLT;                // (B,4,E,L)
constexpr size_t OFF_YS  = OFF_YB  + SZ_DLT;                // (B,E,L)
constexpr size_t OFF_U   = OFF_YS  + SZ_XI;                 // (B,E,L)
constexpr size_t OFF_T1  = OFF_U   + SZ_XI;                 // (B,2E,L)
constexpr size_t OFF_T2  = OFF_T1  + SZ_XZ;                 // (B,2E,L)
constexpr size_t OFF_T3  = OFF_T2  + SZ_XZ;                 // (B,E/2,L)
constexpr size_t OFF_POOL = OFF_T3 + (size_t)B_ * EH_ * L_;
constexpr size_t OFF_GATE = OFF_POOL + (size_t)B_ * EH_;
constexpr size_t WS_FLOATS = OFF_GATE + (size_t)B_ * EH_;   // ~16.4M f32 = 65.5 MB

// Scan-order permutation: xs[k][l] = xi[permL(k,l)]; identical map scatters y back.
DEV int permL(int k, int l) {
  switch (k & 3) {
    case 0:  return l;
    case 1:  return L_ - 1 - l;
    case 2:  return (l % H_) * W_ + (l / H_);
    default: { int r = L_ - 1 - l; return (r % H_) * W_ + (r / H_); }
  }
}

// Copy one 32(K) x 16(N) f32 tile (row stride L_) into LDS.
// 2 KB = 4 issues x 32 lanes x 16 B.  Async (ASYNCcnt) when available.
DEV void tile_to_lds(const float* __restrict__ gbase, float* lds_tile, int lane) {
#pragma unroll
  for (int i = 0; i < 4; ++i) {
    int c = i * 32 + lane;                  // 128 16-byte chunks
    int row = c >> 2, col4 = (c & 3) * 4;
    const float* gp = gbase + (size_t)row * L_ + col4;
    float* lp = lds_tile + row * 16 + col4;
#if ASYNC_LDS_OK
    __builtin_amdgcn_global_load_async_to_lds_b128(
        (GLOBAL_AS i32x4*)gp, (LDS_AS i32x4*)lp, 0, 0);
#else
    *(f32x4*)lp = *(const f32x4*)gp;
#endif
  }
}

DEV void wait_async_lds() {
#if ASYNC_LDS_OK
#  if __has_builtin(__builtin_amdgcn_s_wait_asynccnt)
  __builtin_amdgcn_s_wait_asynccnt(0);
#  else
  asm volatile("s_wait_asynccnt 0x0" ::: "memory");
#  endif
#endif
}

// ---------------- K1: in_proj GEMM  xz[b][o][l] = sum_c W[o][c]*x[b][c][l] ---
__global__ void k_in_proj(const float* __restrict__ x, const float* __restrict__ Wp,
                          float* __restrict__ xz) {
  const int lane = threadIdx.x;
  const int mn = lane & 15, hi = lane >> 4;
  const int n0 = blockIdx.x * 16, m0 = blockIdx.y * 16, b = blockIdx.z;
  const float* xb = x + (size_t)b * DM_ * L_;
  v8f acc = {};
  for (int k0 = 0; k0 < DM_; k0 += 32) {
    v16h af, bf;
    const float* wrow = Wp + (size_t)(m0 + mn) * DM_ + k0 + 16 * hi;
#pragma unroll
    for (int j = 0; j < 8; ++j) { af[j] = (_Float16)wrow[j]; af[8 + j] = (_Float16)wrow[8 + j]; }
    const float* bcol = xb + (size_t)(k0 + 16 * hi) * L_ + n0 + mn;
#pragma unroll
    for (int j = 0; j < 16; ++j) bf[j] = (_Float16)bcol[(size_t)j * L_];
    acc = __builtin_amdgcn_wmma_f32_16x16x32_f16(false, af, false, bf, (short)0, acc, false, false);
  }
  float* o = xz + ((size_t)b * E2_ + m0) * L_ + n0 + mn;
#pragma unroll
  for (int r = 0; r < 8; ++r) o[(size_t)(r + 8 * hi) * L_] = acc[r];
}

// ---------------- K2: depthwise 3x3 + SiLU on channels [0,E) of xz ----------
__global__ void k_dwconv_silu(const float* __restrict__ xz, const float* __restrict__ cw,
                              const float* __restrict__ cb, float* __restrict__ xi) {
  int idx = blockIdx.x * blockDim.x + threadIdx.x;
  if (idx >= B_ * E_ * L_) return;
  int l = idx % L_; int e = (idx / L_) % E_; int b = idx / (L_ * E_);
  int h = l / W_, w = l % W_;
  const float* src = xz + ((size_t)b * E2_ + e) * L_;
  const float* wk = cw + e * 9;
  float acc = cb[e];
#pragma unroll
  for (int dh = -1; dh <= 1; ++dh)
#pragma unroll
    for (int dw = -1; dw <= 1; ++dw) {
      int hh = h + dh, ww = w + dw;
      if (hh >= 0 && hh < H_ && ww >= 0 && ww < W_)
        acc += src[hh * W_ + ww] * wk[(dh + 1) * 3 + (dw + 1)];
    }
  xi[idx] = acc / (1.f + __expf(-acc));     // silu
}

// ---------------- K3: x_proj GEMM with fused direction gather ---------------
__global__ void k_xproj(const float* __restrict__ xi, const float* __restrict__ xpw,
                        float* __restrict__ dbc) {
  const int lane = threadIdx.x;
  const int mn = lane & 15, hi = lane >> 4;
  const int n0 = blockIdx.x * 16, m0 = blockIdx.y * 16;
  const int k = blockIdx.z & 3, b = blockIdx.z >> 2;
  const float* Wk = xpw + (size_t)k * DROWS_ * E_;
  const float* xb = xi + (size_t)b * E_ * L_;
  const int col = permL(k, n0 + mn);
  const bool mv = (m0 + mn) < DROWS_;
  v8f acc = {};
  for (int k0 = 0; k0 < E_; k0 += 32) {
    v16h af, bf;
    const float* wrow = Wk + (size_t)(m0 + mn) * E_ + k0 + 16 * hi;
#pragma unroll
    for (int j = 0; j < 8; ++j) {
      af[j]     = mv ? (_Float16)wrow[j]      : (_Float16)0.f;
      af[8 + j] = mv ? (_Float16)wrow[8 + j] : (_Float16)0.f;
    }
    const float* bcol = xb + (size_t)(k0 + 16 * hi) * L_ + col;
#pragma unroll
    for (int j = 0; j < 16; ++j) bf[j] = (_Float16)bcol[(size_t)j * L_];
    acc = __builtin_amdgcn_wmma_f32_16x16x32_f16(false, af, false, bf, (short)0, acc, false, false);
  }
  float* o = dbc + (((size_t)(b * 4 + k)) * 48 + m0) * L_ + n0 + mn;
#pragma unroll
  for (int r = 0; r < 8; ++r) o[(size_t)(r + 8 * hi) * L_] = acc[r];
}

// ---------------- K4: dt_proj (K=6) + softplus -------------------------------
__global__ void k_dt(const float* __restrict__ dbc, const float* __restrict__ dtw,
                     const float* __restrict__ dtb, float* __restrict__ dlt) {
  int idx = blockIdx.x * blockDim.x + threadIdx.x;
  if (idx >= B_ * 4 * E_ * L_) return;
  int l = idx % L_; int d = (idx / L_) % E_;
  int k = (idx / (L_ * E_)) & 3; int b = idx / (L_ * E_ * 4);
  const float* row = dbc + ((size_t)(b * 4 + k) * 48) * L_ + l;   // rows 0..5 = dlt
  const float* wv = dtw + ((size_t)k * E_ + d) * R_;
  float a = dtb[k * E_ + d];
#pragma unroll
  for (int r = 0; r < R_; ++r) a += wv[r] * row[(size_t)r * L_];
  dlt[idx] = (a > 20.f) ? a : log1pf(__expf(a));                  // softplus
}

// ---------------- K5: selective scan, 16 states in registers ----------------
__global__ void k_scan(const float* __restrict__ xi, const float* __restrict__ dbc,
                       const float* __restrict__ dlt, const float* __restrict__ A_log,
                       const float* __restrict__ Ds, float* __restrict__ ybuf) {
  int idx = blockIdx.x * blockDim.x + threadIdx.x;
  if (idx >= B_ * 4 * E_) return;
  int d = idx % E_; int k = (idx / E_) & 3; int b = idx / (E_ * 4);
  float Areg[N_], h[N_];
#pragma unroll
  for (int n = 0; n < N_; ++n) { Areg[n] = A_log[((size_t)k * E_ + d) * N_ + n]; h[n] = 0.f; }
  const float* bptr = dbc + ((size_t)(b * 4 + k) * 48 + R_) * L_;        // rows 6..21
  const float* cptr = dbc + ((size_t)(b * 4 + k) * 48 + R_ + N_) * L_;   // rows 22..37
  const float* dptr = dlt + ((size_t)(b * 4 + k) * E_ + d) * L_;
  const float* xb   = xi  + ((size_t)b * E_ + d) * L_;
  const float Dv = Ds[k * E_ + d];
  float* yo = ybuf + ((size_t)(b * 4 + k) * E_ + d) * L_;
  for (int l = 0; l < L_; ++l) {
    if ((l & 63) == 0 && l + 256 < L_) {               // cover HBM latency:
      __builtin_prefetch(dptr + l + 256, 0, 0);        // -> global_prefetch_b8
      __builtin_prefetch(bptr + l + 256, 0, 0);
      __builtin_prefetch(cptr + l + 256, 0, 0);
    }
    float dt = dptr[l];
    float xv = xb[permL(k, l)];
    float dx = dt * xv;
    float y = Dv;
#pragma unroll
    for (int n = 0; n < N_; ++n) {
      float a = __expf(-dt * Areg[n]);
      h[n] = a * h[n] + dx * bptr[(size_t)n * L_ + l];
      y += h[n] * cptr[(size_t)n * L_ + l];
    }
    yo[l] = y;
  }
}

// ---------------- K6: combine 4 directions back to raster order -------------
__global__ void k_combine(const float* __restrict__ yb, float* __restrict__ ys) {
  int idx = blockIdx.x * blockDim.x + threadIdx.x;
  if (idx >= B_ * E_ * L_) return;
  int q = idx % L_; int d = (idx / L_) % E_; int b = idx / (L_ * E_);
  int p2 = (q % H_) * W_ + (q / H_);
  const float* base = yb + (size_t)b * 4 * E_ * L_ + (size_t)d * L_;
  const size_t ks = (size_t)E_ * L_;
  ys[idx] = base[q] + base[ks + (L_ - 1 - q)] + base[2 * ks + p2] + base[3 * ks + (L_ - 1 - p2)];
}

// ---------------- K7: channel LayerNorm + gate by z -------------------------
__global__ void k_ln_gate(const float* __restrict__ ys, const float* __restrict__ xz,
                          const float* __restrict__ g, const float* __restrict__ be,
                          float* __restrict__ u) {
  int idx = blockIdx.x * blockDim.x + threadIdx.x;
  if (idx >= B_ * L_) return;
  int q = idx % L_, b = idx / L_;
  const float* yp = ys + (size_t)b * E_ * L_ + q;
  float s = 0.f, s2 = 0.f;
  for (int d = 0; d < E_; ++d) { float v = yp[(size_t)d * L_]; s += v; s2 += v * v; }
  float mu = s / (float)E_;
  float var = s2 / (float)E_ - mu * mu;
  float inv = rsqrtf(var + 1e-5f);
  const float* zp = xz + ((size_t)b * E2_ + E_) * L_ + q;     // z = channels [E,2E)
  float* up = u + (size_t)b * E_ * L_ + q;
  for (int d = 0; d < E_; ++d) {
    float v = (yp[(size_t)d * L_] - mu) * inv * g[d] + be[d];
    up[(size_t)d * L_] = v * zp[(size_t)d * L_];
  }
}

// ---------------- K8: c1 1x1 GEMM + BN + ReLU, async-LDS staged B -----------
__global__ void k_c1(const float* __restrict__ u, const float* __restrict__ Wp,
                     const float* __restrict__ bias, const float* __restrict__ g,
                     const float* __restrict__ bb, const float* __restrict__ mm,
                     const float* __restrict__ vv, float* __restrict__ t1) {
  __shared__ float bt[2][32 * 16];                 // double-buffered B tile
  const int lane = threadIdx.x;
  const int mn = lane & 15, hi = lane >> 4;
  const int n0 = blockIdx.x * 16, m0 = blockIdx.y * 16, b = blockIdx.z;
  const float* ub = u + (size_t)b * E_ * L_ + n0;
  tile_to_lds(ub, bt[0], lane);                    // prologue: stage k0=0 tile
  v8f acc = {};
  for (int k0 = 0, it = 0; k0 < E_; k0 += 32, ++it) {
    v16h af, bf;
    const float* wrow = Wp + (size_t)(m0 + mn) * E_ + k0 + 16 * hi;
#pragma unroll
    for (int j = 0; j < 8; ++j) { af[j] = (_Float16)wrow[j]; af[8 + j] = (_Float16)wrow[8 + j]; }
    wait_async_lds();                              // tile (it&1) resident
    if (k0 + 32 < E_)
      tile_to_lds(ub + (size_t)(k0 + 32) * L_, bt[(it + 1) & 1], lane);
    const float* tp = bt[it & 1] + 16 * 16 * hi + mn;
#pragma unroll
    for (int j = 0; j < 16; ++j) bf[j] = (_Float16)tp[j * 16];
    acc = __builtin_amdgcn_wmma_f32_16x16x32_f16(false, af, false, bf, (short)0, acc, false, false);
  }
  float* o = t1 + ((size_t)b * E2_ + m0) * L_ + n0 + mn;
#pragma unroll
  for (int r = 0; r < 8; ++r) {
    int oc = m0 + r + 8 * hi;
    float val = acc[r] + bias[oc];
    val = (val - mm[oc]) * rsqrtf(vv[oc] + 1e-5f) * g[oc] + bb[oc];
    o[(size_t)(r + 8 * hi) * L_] = fmaxf(val, 0.f);
  }
}

// ---------------- K9: depthwise 3x3 on 2E channels + BN + ReLU --------------
__global__ void k_dwconv2(const float* __restrict__ t1, const float* __restrict__ cw,
                          const float* __restrict__ cb, const float* __restrict__ g,
                          const float* __restrict__ bb, const float* __restrict__ mm,
                          const float* __restrict__ vv, float* __restrict__ t2) {
  int idx = blockIdx.x * blockDim.x + threadIdx.x;
  if (idx >= B_ * E2_ * L_) return;
  int l = idx % L_; int c = (idx / L_) % E2_; int b = idx / (L_ * E2_);
  int h = l / W_, w = l % W_;
  const float* src = t1 + ((size_t)b * E2_ + c) * L_;
  const float* wk = cw + c * 9;
  float acc = cb[c];
#pragma unroll
  for (int dh = -1; dh <= 1; ++dh)
#pragma unroll
    for (int dw = -1; dw <= 1; ++dw) {
      int hh = h + dh, ww = w + dw;
      if (hh >= 0 && hh < H_ && ww >= 0 && ww < W_)
        acc += src[hh * W_ + ww] * wk[(dh + 1) * 3 + (dw + 1)];
    }
  float val = (acc - mm[c]) * rsqrtf(vv[c] + 1e-5f) * g[c] + bb[c];
  t2[idx] = fmaxf(val, 0.f);
}

// ---------------- K10: c3 1x1 GEMM (2E -> E/2) + bias, async-LDS staged B ---
__global__ void k_c3(const float* __restrict__ t2, const float* __restrict__ Wp,
                     const float* __restrict__ bias, float* __restrict__ t3) {
  __shared__ float bt[2][32 * 16];
  const int lane = threadIdx.x;
  const int mn = lane & 15, hi = lane >> 4;
  const int n0 = blockIdx.x * 16, m0 = blockIdx.y * 16, b = blockIdx.z;
  const float* tb = t2 + (size_t)b * E2_ * L_ + n0;
  tile_to_lds(tb, bt[0], lane);
  v8f acc = {};
  for (int k0 = 0, it = 0; k0 < E2_; k0 += 32, ++it) {
    v16h af, bf;
    const float* wrow = Wp + (size_t)(m0 + mn) * E2_ + k0 + 16 * hi;
#pragma unroll
    for (int j = 0; j < 8; ++j) { af[j] = (_Float16)wrow[j]; af[8 + j] = (_Float16)wrow[8 + j]; }
    wait_async_lds();
    if (k0 + 32 < E2_)
      tile_to_lds(tb + (size_t)(k0 + 32) * L_, bt[(it + 1) & 1], lane);
    const float* tp = bt[it & 1] + 16 * 16 * hi + mn;
#pragma unroll
    for (int j = 0; j < 16; ++j) bf[j] = (_Float16)tp[j * 16];
    acc = __builtin_amdgcn_wmma_f32_16x16x32_f16(false, af, false, bf, (short)0, acc, false, false);
  }
  float* o = t3 + ((size_t)b * EH_ + m0) * L_ + n0 + mn;
#pragma unroll
  for (int r = 0; r < 8; ++r) o[(size_t)(r + 8 * hi) * L_] = acc[r] + bias[m0 + r + 8 * hi];
}

// ---------------- K11: SE global average pool -------------------------------
__global__ void k_pool(const float* __restrict__ t3, float* __restrict__ pool) {
  __shared__ float sm[256];
  const int bm = blockIdx.x;                      // b*EH + m
  const float* p = t3 + (size_t)bm * L_;
  float s = 0.f;
  for (int l = threadIdx.x; l < L_; l += 256) s += p[l];
  sm[threadIdx.x] = s; __syncthreads();
  for (int o = 128; o > 0; o >>= 1) {
    if ((int)threadIdx.x < o) sm[threadIdx.x] += sm[threadIdx.x + o];
    __syncthreads();
  }
  if (threadIdx.x == 0) pool[bm] = sm[0] / (float)L_;
}

// ---------------- K12: SE two tiny FCs + sigmoid ----------------------------
__global__ void k_se(const float* __restrict__ pool, const float* __restrict__ w1,
                     const float* __restrict__ b1, const float* __restrict__ w2,
                     const float* __restrict__ b2, float* __restrict__ gate) {
  __shared__ float hid[EQ_];
  const int b = blockIdx.x, t = threadIdx.x;
  if (t < EQ_) {
    float a = b1[t];
    for (int j = 0; j < EH_; ++j) a += w1[t * EH_ + j] * pool[b * EH_ + j];
    hid[t] = fmaxf(a, 0.f);
  }
  __syncthreads();
  float a = b2[t];
  for (int j = 0; j < EQ_; ++j) a += w2[t * EQ_ + j] * hid[j];
  gate[b * EH_ + t] = 1.f / (1.f + __expf(-a));
}

// ---------------- K13: out = t3 * (1 + gate) --------------------------------
__global__ void k_apply(const float* __restrict__ t3, const float* __restrict__ gate,
                        float* __restrict__ out) {
  int idx = blockIdx.x * blockDim.x + threadIdx.x;
  if (idx >= B_ * EH_ * L_) return;
  int m = (idx / L_) % EH_; int b = idx / (L_ * EH_);
  out[idx] = t3[idx] * (1.f + gate[b * EH_ + m]);
}

// ---------------------------------------------------------------------------
extern "C" void kernel_launch(void* const* d_in, const int* in_sizes, int n_in,
                              void* d_out, int out_size, void* d_ws, size_t ws_size,
                              hipStream_t stream) {
  (void)in_sizes; (void)n_in; (void)out_size;
  if (ws_size < WS_FLOATS * sizeof(float)) return;

  const float* x         = (const float*)d_in[0];
  const float* in_proj_w = (const float*)d_in[1];
  const float* conv_w    = (const float*)d_in[2];
  const float* conv_b    = (const float*)d_in[3];
  const float* x_proj_w  = (const float*)d_in[4];
  const float* dt_proj_w = (const float*)d_in[5];
  const float* dt_proj_b = (const float*)d_in[6];
  const float* A_log     = (const float*)d_in[7];
  const float* Ds        = (const float*)d_in[8];
  const float* ln_g      = (const float*)d_in[9];
  const float* ln_b      = (const float*)d_in[10];
  const float* c1_w      = (const float*)d_in[11];
  const float* c1_b      = (const float*)d_in[12];
  const float* bn1_g     = (const float*)d_in[13];
  const float* bn1_b     = (const float*)d_in[14];
  const float* bn1_m     = (const float*)d_in[15];
  const float* bn1_v     = (const float*)d_in[16];
  const float* c2_w      = (const float*)d_in[17];
  const float* c2_b      = (const float*)d_in[18];
  const float* bn2_g     = (const float*)d_in[19];
  const float* bn2_b     = (const float*)d_in[20];
  const float* bn2_m     = (const float*)d_in[21];
  const float* bn2_v     = (const float*)d_in[22];
  const float* c3_w      = (const float*)d_in[23];
  const float* c3_b      = (const float*)d_in[24];
  const float* se1_w     = (const float*)d_in[25];
  const float* se1_b     = (const float*)d_in[26];
  const float* se2_w     = (const float*)d_in[27];
  const float* se2_b     = (const float*)d_in[28];

  float* ws   = (float*)d_ws;
  float* xz   = ws + OFF_XZ;
  float* xi   = ws + OFF_XI;
  float* dbc  = ws + OFF_DBC;
  float* dlt  = ws + OFF_DLT;
  float* yb   = ws + OFF_YB;
  float* ys   = ws + OFF_YS;
  float* u    = ws + OFF_U;
  float* t1   = ws + OFF_T1;
  float* t2   = ws + OFF_T2;
  float* t3   = ws + OFF_T3;
  float* pool = ws + OFF_POOL;
  float* gate = ws + OFF_GATE;

  const dim3 wv(32);
  k_in_proj   <<<dim3(L_ / 16, E2_ / 16, B_), wv, 0, stream>>>(x, in_proj_w, xz);
  k_dwconv_silu<<<(B_ * E_ * L_ + 255) / 256, 256, 0, stream>>>(xz, conv_w, conv_b, xi);
  k_xproj     <<<dim3(L_ / 16, 3, B_ * 4),    wv, 0, stream>>>(xi, x_proj_w, dbc);
  k_dt        <<<(B_ * 4 * E_ * L_ + 255) / 256, 256, 0, stream>>>(dbc, dt_proj_w, dt_proj_b, dlt);
  k_scan      <<<(B_ * 4 * E_ + 127) / 128, 128, 0, stream>>>(xi, dbc, dlt, A_log, Ds, yb);
  k_combine   <<<(B_ * E_ * L_ + 255) / 256, 256, 0, stream>>>(yb, ys);
  k_ln_gate   <<<(B_ * L_ + 255) / 256, 256, 0, stream>>>(ys, xz, ln_g, ln_b, u);
  k_c1        <<<dim3(L_ / 16, E2_ / 16, B_), wv, 0, stream>>>(u, c1_w, c1_b, bn1_g, bn1_b, bn1_m, bn1_v, t1);
  k_dwconv2   <<<(B_ * E2_ * L_ + 255) / 256, 256, 0, stream>>>(t1, c2_w, c2_b, bn2_g, bn2_b, bn2_m, bn2_v, t2);
  k_c3        <<<dim3(L_ / 16, EH_ / 16, B_), wv, 0, stream>>>(t2, c3_w, c3_b, t3);
  k_pool      <<<B_ * EH_, 256, 0, stream>>>(t3, pool);
  k_se        <<<B_, EH_, 0, stream>>>(pool, se1_w, se1_b, se2_w, se2_b, gate);
  k_apply     <<<(B_ * EH_ * L_ + 255) / 256, 256, 0, stream>>>(t3, gate, (float*)d_out);
}
